// FPModule_51762945851726
// MI455X (gfx1250) — matrix operand order repaired and checked by reading, not compiled, not run on hardware
//
#include <hip/hip_runtime.h>
#include <hip/hip_bf16.h>

typedef __bf16 bf16_t;
typedef __attribute__((ext_vector_type(16))) __bf16 v16bf;
typedef __attribute__((ext_vector_type(8)))  __bf16 v8bf;
typedef __attribute__((ext_vector_type(8)))  float   v8f;
typedef __attribute__((ext_vector_type(4)))  float   v4f;

#define BB        4
#define N_COARSE  1024
#define M_FINE    4096
#define MTOT      (BB * M_FINE)     // 16384 rows
#define C_IN      512
#define C_SKIP    256
#define KDIM1     (C_IN + C_SKIP)   // 768
#define C_HID     512
#define C_OUT     512

// ---------------------------------------------------------------------------
// KNN interpolation: one wave (32 lanes) per query point.
// ---------------------------------------------------------------------------
__global__ __launch_bounds__(256)
void knn_interp_kernel(const float* __restrict__ x,
                       const float* __restrict__ pos,
                       const float* __restrict__ pos_skip,
                       bf16_t* __restrict__ hin)
{
    const int lane = threadIdx.x & 31;
    const int wave = threadIdx.x >> 5;
    const int q    = blockIdx.x * 8 + wave;        // 0..16383
    const int b    = q >> 12;                      // M_FINE = 4096 = 2^12

    const float qx = pos_skip[q * 3 + 0];
    const float qy = pos_skip[q * 3 + 1];
    const float qz = pos_skip[q * 3 + 2];

    float t0 = 3.4e38f, t1 = 3.4e38f, t2 = 3.4e38f;
    int   j0 = 0, j1 = 0, j2 = 0;

    const float* __restrict__ pb = pos + (size_t)b * N_COARSE * 3;
    for (int i = lane; i < N_COARSE; i += 32) {
        float dx = qx - pb[i * 3 + 0];
        float dy = qy - pb[i * 3 + 1];
        float dz = qz - pb[i * 3 + 2];
        float d2 = dx * dx + dy * dy + dz * dz;
        if (d2 < t2) {
            if (d2 < t1) {
                t2 = t1; j2 = j1;
                if (d2 < t0) { t1 = t0; j1 = j0; t0 = d2; j0 = i; }
                else         { t1 = d2; j1 = i; }
            } else { t2 = d2; j2 = i; }
        }
    }

    // wave-wide merge of sorted triples
    for (int s = 16; s > 0; s >>= 1) {
        float o0 = __shfl_xor(t0, s, 32);
        float o1 = __shfl_xor(t1, s, 32);
        float o2 = __shfl_xor(t2, s, 32);
        int   p0 = __shfl_xor(j0, s, 32);
        int   p1 = __shfl_xor(j1, s, 32);
        int   p2 = __shfl_xor(j2, s, 32);
        float n0, n1, n2; int m0, m1, m2;
        if (t0 <= o0) {
            n0 = t0; m0 = j0;
            if (t1 <= o0) {
                n1 = t1; m1 = j1;
                if (t2 <= o0) { n2 = t2; m2 = j2; } else { n2 = o0; m2 = p0; }
            } else {
                n1 = o0; m1 = p0;
                if (t1 <= o1) { n2 = t1; m2 = j1; } else { n2 = o1; m2 = p1; }
            }
        } else {
            n0 = o0; m0 = p0;
            if (o1 <= t0) {
                n1 = o1; m1 = p1;
                if (o2 <= t0) { n2 = o2; m2 = p2; } else { n2 = t0; m2 = j0; }
            } else {
                n1 = t0; m1 = j0;
                if (t1 <= o1) { n2 = t1; m2 = j1; } else { n2 = o1; m2 = p1; }
            }
        }
        t0 = n0; t1 = n1; t2 = n2;
        j0 = m0; j1 = m1; j2 = m2;
    }

    float w0 = 1.0f / fmaxf(t0, 1e-16f);
    float w1 = 1.0f / fmaxf(t1, 1e-16f);
    float w2 = 1.0f / fmaxf(t2, 1e-16f);
    const float inv = 1.0f / (w0 + w1 + w2);
    w0 *= inv; w1 *= inv; w2 *= inv;

    const v4f* __restrict__ r0 = (const v4f*)(x + ((size_t)(b * N_COARSE + j0)) * C_IN + lane * 16);
    const v4f* __restrict__ r1 = (const v4f*)(x + ((size_t)(b * N_COARSE + j1)) * C_IN + lane * 16);
    const v4f* __restrict__ r2 = (const v4f*)(x + ((size_t)(b * N_COARSE + j2)) * C_IN + lane * 16);

    bf16_t vals[16];
#pragma unroll
    for (int t = 0; t < 4; ++t) {
        v4f a = r0[t], c = r1[t], d = r2[t];
#pragma unroll
        for (int e = 0; e < 4; ++e)
            vals[t * 4 + e] = (bf16_t)(w0 * a[e] + w1 * c[e] + w2 * d[e]);
    }
    bf16_t* __restrict__ op = hin + (size_t)q * KDIM1 + lane * 16;
    v8bf s0, s1;
#pragma unroll
    for (int e = 0; e < 8; ++e) { s0[e] = vals[e]; s1[e] = vals[8 + e]; }
    *(v8bf*)(op)     = s0;
    *(v8bf*)(op + 8) = s1;
}

// copy x_skip (f32, 16384x256) -> bf16 cols [512,768) of hin
__global__ __launch_bounds__(256)
void skip_to_hin(const float* __restrict__ xs, bf16_t* __restrict__ hin)
{
    const int idx = blockIdx.x * 256 + threadIdx.x;   // < 16384*256
    const int m = idx >> 8;
    const int c = idx & 255;
    hin[(size_t)m * KDIM1 + C_IN + c] = (bf16_t)xs[idx];
}

// W (K x 512, f32, row-major) -> Wt (512 x K, bf16, row-major)
__global__ __launch_bounds__(256)
void conv_transpose_w(const float* __restrict__ W, bf16_t* __restrict__ Wt, int Kd)
{
    const int idx = blockIdx.x * 256 + threadIdx.x;   // < Kd*512
    const int k = idx >> 9;
    const int n = idx & 511;
    Wt[(size_t)n * Kd + k] = (bf16_t)W[idx];
}

// ---------------------------------------------------------------------------
// bf16 WMMA GEMM, async double-buffered LDS staging, bank-conflict-free:
//   C[M x 512] = act(A[M x KDIM] @ Bt^T + bias)
//   A  : bf16 row-major M x KDIM;  Bt : bf16 row-major 512 x KDIM
// Block = 8 waves = 128x128 tile. Per 32-wide K step the block stages a
// 128x32 A chunk and B chunk in LDS via global_load_async_to_lds_b128
// (ASYNCcnt, double buffered). LDS rows are padded to 80 B (20 banks) so
// every ds_load_b128 hits each 4-bank group exactly twice (optimal).
// LDS: 4 x 10240 B = 40 KB.
// ---------------------------------------------------------------------------
#define ROW_B   80          // padded LDS row stride in bytes (32 data + pad)
#define CHUNK_B (128 * ROW_B)   // 10240 B per chunk buffer

template <int KDIM, bool RELU, typename OutT>
__global__ __launch_bounds__(256)
void gemm_wmma(const bf16_t* __restrict__ A,
               const bf16_t* __restrict__ Bt,
               const float*  __restrict__ bias,
               OutT* __restrict__ C)
{
    constexpr int KSTEPS = KDIM / 32;
    // bytes: A stage s at s*CHUNK_B ; B stage s at 2*CHUNK_B + s*CHUNK_B
    __shared__ __align__(16) char smem[4 * CHUNK_B];   // 40 KB

    const int tid   = threadIdx.x;
    const int lane  = tid & 31;
    const int wave  = tid >> 5;
    const int waveM = wave & 1;        // 2 wave-rows of 64
    const int waveN = wave >> 1;       // 4 wave-cols of 32
    const int hlf   = lane >> 4;
    const int l16   = lane & 15;

    const int rowBlock = blockIdx.x * 128;
    const int colBlock = blockIdx.y * 128;

    // cooperative async-copy mapping: thread t copies 32 B = 2 x b128
    const int crow  = tid >> 1;
    const int cpart = tid & 1;
    const bf16_t* gA = A  + (size_t)(rowBlock + crow) * KDIM + cpart * 16;
    const bf16_t* gB = Bt + (size_t)(colBlock + crow) * KDIM + cpart * 16;

    // low 32 bits of an LDS flat address == LDS byte offset (ISA 10.2)
    const unsigned ldsBase = (unsigned)(uintptr_t)(void*)&smem[0];
    const unsigned ldsA    = ldsBase + (unsigned)(crow * ROW_B + cpart * 32);
    const unsigned ldsB    = ldsA + 2u * CHUNK_B;

#define ISSUE_CHUNK(kc_, stage_)                                              \
    do {                                                                      \
        unsigned long long ga_ = (unsigned long long)(uintptr_t)(gA + (kc_) * 32); \
        unsigned long long gb_ = (unsigned long long)(uintptr_t)(gB + (kc_) * 32); \
        unsigned la_ = ldsA + (unsigned)((stage_) * CHUNK_B);                 \
        unsigned lb_ = ldsB + (unsigned)((stage_) * CHUNK_B);                 \
        asm volatile("global_load_async_to_lds_b128 %0, %1, off\n\t"          \
                     "global_load_async_to_lds_b128 %0, %1, off offset:16"    \
                     :: "v"(la_), "v"(ga_) : "memory");                       \
        asm volatile("global_load_async_to_lds_b128 %0, %1, off\n\t"          \
                     "global_load_async_to_lds_b128 %0, %1, off offset:16"    \
                     :: "v"(lb_), "v"(gb_) : "memory");                       \
    } while (0)

    v8f acc[4][2];
#pragma unroll
    for (int mi = 0; mi < 4; ++mi)
#pragma unroll
        for (int ni = 0; ni < 2; ++ni)
            acc[mi][ni] = (v8f){0.f, 0.f, 0.f, 0.f, 0.f, 0.f, 0.f, 0.f};

    auto compute_stage = [&](int stage) {
        const int aS = stage * CHUNK_B;
        const int bS = 2 * CHUNK_B + stage * CHUNK_B;
        v16bf afr[4], bfr[2];
#pragma unroll
        for (int mi = 0; mi < 4; ++mi) {
            const int e = aS + (waveM * 64 + mi * 16 + l16) * ROW_B + hlf * 16;
            v8bf lo = *(const v8bf*)(smem + e);        // K = hlf*8 .. +7
            v8bf hi = *(const v8bf*)(smem + e + 32);   // K = 16 + hlf*8 .. +7
#pragma unroll
            for (int t = 0; t < 8; ++t) { afr[mi][t] = lo[t]; afr[mi][8 + t] = hi[t]; }
        }
#pragma unroll
        for (int ni = 0; ni < 2; ++ni) {
            const int e = bS + (waveN * 32 + ni * 16 + l16) * ROW_B + hlf * 16;
            v8bf lo = *(const v8bf*)(smem + e);        // K = hlf*8 .. +7
            v8bf hi = *(const v8bf*)(smem + e + 32);   // K = 16 + hlf*8 .. +7
#pragma unroll
            for (int t = 0; t < 8; ++t) { bfr[ni][t] = lo[t]; bfr[ni][8 + t] = hi[t]; }
        }
#pragma unroll
        for (int mi = 0; mi < 4; ++mi)
#pragma unroll
            for (int ni = 0; ni < 2; ++ni)
                acc[mi][ni] = __builtin_amdgcn_wmma_f32_16x16x32_bf16(
                    false, afr[mi], false, bfr[ni], (short)0, acc[mi][ni], false, false);
    };

    ISSUE_CHUNK(0, 0);

    // steady state: straight-line hot loop, no wait-immediate branching
    for (int kc = 0; kc < KSTEPS - 1; ++kc) {
        ISSUE_CHUNK(kc + 1, (kc + 1) & 1);
        // <=8 async ops outstanding; in-order completion -> waiting to <=4
        // guarantees the current stage's 4 ops are done.
        asm volatile("s_wait_asynccnt 0x4" ::: "memory");
        __syncthreads();                 // all waves' copies visible
        compute_stage(kc & 1);
        __syncthreads();                 // reads done before refill of this buf
    }
    // peeled final step
    asm volatile("s_wait_asynccnt 0x0" ::: "memory");
    __syncthreads();
    compute_stage((KSTEPS - 1) & 1);
#undef ISSUE_CHUNK

#pragma unroll
    for (int mi = 0; mi < 4; ++mi)
#pragma unroll
        for (int ni = 0; ni < 2; ++ni) {
            const int n  = colBlock + waveN * 32 + ni * 16 + l16;
            const float bs = bias[n];
#pragma unroll
            for (int r = 0; r < 8; ++r) {
                const int m = rowBlock + waveM * 64 + mi * 16 + r + hlf * 8;
                float v = acc[mi][ni][r] + bs;
                if (RELU) v = fmaxf(v, 0.0f);
                C[(size_t)m * 512 + n] = (OutT)v;
            }
        }
}

// ---------------------------------------------------------------------------
extern "C" void kernel_launch(void* const* d_in, const int* in_sizes, int n_in,
                              void* d_out, int out_size, void* d_ws, size_t ws_size,
                              hipStream_t stream)
{
    (void)in_sizes; (void)n_in; (void)ws_size;

    const float* x          = (const float*)d_in[0];   // 4096 x 512
    const float* pos        = (const float*)d_in[1];   // 4096 x 3
    const float* x_skip     = (const float*)d_in[3];   // 16384 x 256
    const float* pos_skip   = (const float*)d_in[4];   // 16384 x 3
    const void*  batch_skip = d_in[5];                 // 16384 (int64)
    const float* W1         = (const float*)d_in[6];   // 768 x 512
    const float* b1         = (const float*)d_in[7];   // 512
    const float* W2         = (const float*)d_in[8];   // 512 x 512
    const float* b2         = (const float*)d_in[9];   // 512

    // workspace layout (16B-aligned sections)
    char* ws = (char*)d_ws;
    bf16_t* hin = (bf16_t*)ws;                                      // 16384 x 768 bf16 (25,165,824 B)
    bf16_t* w1t = (bf16_t*)(ws + 25165824);                         //   512 x 768 bf16 (   786,432 B)
    bf16_t* h   = (bf16_t*)(ws + 25165824 + 786432);                // 16384 x 512 bf16 (16,777,216 B)
    bf16_t* w2t = (bf16_t*)(ws + 25165824 + 786432 + 16777216);     //   512 x 512 bf16 (   524,288 B)
    float*  out = (float*)d_out;

    // 1) KNN interpolation -> hin[:, 0:512]
    knn_interp_kernel<<<MTOT / 8, 256, 0, stream>>>(x, pos, pos_skip, hin);
    // 2) x_skip -> hin[:, 512:768]
    skip_to_hin<<<(MTOT * C_SKIP) / 256, 256, 0, stream>>>(x_skip, hin);
    // 3) weight transpose+convert
    conv_transpose_w<<<(KDIM1 * 512) / 256, 256, 0, stream>>>(W1, w1t, KDIM1);
    conv_transpose_w<<<(C_HID * 512) / 256, 256, 0, stream>>>(W2, w2t, C_HID);
    // 4) h = relu(hin @ W1 + b1)   [bf16 out]
    gemm_wmma<KDIM1, true, bf16_t><<<dim3(MTOT / 128, 4), 256, 0, stream>>>(hin, w1t, b1, h);
    // 5) out = h @ W2 + b2         [f32 out]
    gemm_wmma<C_HID, false, float><<<dim3(MTOT / 128, 4), 256, 0, stream>>>(h, w2t, b2, out);

    // 6) tuple tails: pos_skip (f32) then batch_skip (raw bytes, size derived
    //    from out_size so int64 vs int32 packing both work)
    const size_t outElems  = (size_t)MTOT * C_OUT;           // 8,388,608
    const size_t posBytes  = (size_t)MTOT * 3 * sizeof(float);
    hipMemcpyAsync(out + outElems, pos_skip, posBytes, hipMemcpyDeviceToDevice, stream);
    const long long tailBytes = (long long)out_size * 4 - (long long)outElems * 4 - (long long)posBytes;
    if (tailBytes > 0)
        hipMemcpyAsync((char*)d_out + outElems * 4 + posBytes, batch_skip,
                       (size_t)tailBytes, hipMemcpyDeviceToDevice, stream);
}
